// StrokeDecoderGenerator_27556510171516
// MI455X (gfx1250) — matrix operand order repaired and checked by reading, not compiled, not run on hardware
//
#include <hip/hip_runtime.h>
#include <hip/hip_bf16.h>

// Problem constants (from reference): B=512, LAT=128, H=1024, D=5, L=2, T=256
#define BATCH 512
#define HID   1024
#define LATD  128
#define OUTD  5
#define SEQT  256
#define NLAY  2

typedef __attribute__((ext_vector_type(16))) __bf16 v16bf;
typedef __attribute__((ext_vector_type(8)))  __bf16 bf16x8;
typedef __attribute__((ext_vector_type(8)))  float  v8f;

union FragAB { v16bf v; bf16x8 h[2]; };

__device__ __forceinline__ float sigmoidf(float x) { return 1.0f / (1.0f + __expf(-x)); }

#define WMMA_BF16(ACC, AV, BV) \
    ACC = __builtin_amdgcn_wmma_f32_16x16x32_bf16(false, (AV), false, (BV), (short)0, (ACC), false, false)

// ---------------------------------------------------------------------------
// Fused LSTM cell: gates = A1 @ W1^T (+ A2 @ W2^T) (+ x @ Wx^T) + bias
// A1/A2: [BATCH, HID] bf16 (row-major, K contiguous)
// W1/W2: [4*HID, HID] bf16 (row-major, K contiguous) -> B operand
// Wave tile: 32(M) x 16(j) for ALL FOUR gates (two 16x16 M-subtiles share each
// B fragment -> 1.5 b128 loads per WMMA on layer 1). Cell nonlinearity fused.
// Block: 8 waves = 2(M) x 4(J) -> 64M x 64j. Grid: (BATCH/64, HID/64) = (8,16).
// ---------------------------------------------------------------------------
template <bool HAS_A2>
__global__ __launch_bounds__(256)
void lstm_cell_wmma(const __bf16* __restrict__ A1, const __bf16* __restrict__ W1,
                    const __bf16* __restrict__ A2, const __bf16* __restrict__ W2,
                    const float*  __restrict__ xin, int xstride,       // [BATCH, OUTD] or null
                    const float*  __restrict__ Wx,                     // [4*HID, OUTD] f32 or null
                    const float*  __restrict__ bias,                   // [4*HID]
                    const float*  __restrict__ c_in,
                    float*        __restrict__ c_out,
                    __bf16*       __restrict__ h_out)
{
    const int lane  = threadIdx.x & 31;
    const int wave  = threadIdx.x >> 5;
    const int waveM = wave & 1;                  // 2 M-waves
    const int waveJ = wave >> 1;                 // 4 J-waves
    const int l16   = lane & 15;
    const int half  = lane >> 4;                 // 0: lanes 0-15, 1: lanes 16-31
    const int mbase = blockIdx.x * 64 + waveM * 32;
    const int j     = blockIdx.y * 64 + waveJ * 16 + l16;

    v8f acc0a = {0,0,0,0,0,0,0,0}, acc0b = {0,0,0,0,0,0,0,0};
    v8f acc1a = {0,0,0,0,0,0,0,0}, acc1b = {0,0,0,0,0,0,0,0};
    v8f acc2a = {0,0,0,0,0,0,0,0}, acc2b = {0,0,0,0,0,0,0,0};
    v8f acc3a = {0,0,0,0,0,0,0,0}, acc3b = {0,0,0,0,0,0,0,0};

    // A fragment (16x32 bf16) in contiguous k order (same as B):
    //   halves 0..7  = K[k0 + half*16     .. +7]
    //   halves 8..15 = K[k0 + half*16 + 8 .. +7]
    const __bf16* a1lo = A1 + (size_t)(mbase + l16) * HID;
    const __bf16* a1hi = A1 + (size_t)(mbase + 16 + l16) * HID;
    const __bf16* a2lo = nullptr;
    const __bf16* a2hi = nullptr;
    // B fragment (32x16 bf16): lanes 0-15 -> K[k0..k0+15], lanes 16-31 -> K[k0+16..k0+31]
    const __bf16* w1r0 = W1 + (size_t)(0 * HID + j) * HID;
    const __bf16* w1r1 = W1 + (size_t)(1 * HID + j) * HID;
    const __bf16* w1r2 = W1 + (size_t)(2 * HID + j) * HID;
    const __bf16* w1r3 = W1 + (size_t)(3 * HID + j) * HID;
    const __bf16 *w2r0 = nullptr, *w2r1 = nullptr, *w2r2 = nullptr, *w2r3 = nullptr;
    if constexpr (HAS_A2) {
        a2lo = A2 + (size_t)(mbase + l16) * HID;
        a2hi = A2 + (size_t)(mbase + 16 + l16) * HID;
        w2r0 = W2 + (size_t)(0 * HID + j) * HID;
        w2r1 = W2 + (size_t)(1 * HID + j) * HID;
        w2r2 = W2 + (size_t)(2 * HID + j) * HID;
        w2r3 = W2 + (size_t)(3 * HID + j) * HID;
    }

    const int aoff0 = half * 16;
    const int aoff1 = half * 16 + 8;
    const int boff  = half * 16;

    for (int k0 = 0; k0 < HID; k0 += 32) {
        FragAB alo, ahi;
        alo.h[0] = *reinterpret_cast<const bf16x8*>(a1lo + k0 + aoff0);
        alo.h[1] = *reinterpret_cast<const bf16x8*>(a1lo + k0 + aoff1);
        ahi.h[0] = *reinterpret_cast<const bf16x8*>(a1hi + k0 + aoff0);
        ahi.h[1] = *reinterpret_cast<const bf16x8*>(a1hi + k0 + aoff1);

        FragAB b;
        b.h[0] = *reinterpret_cast<const bf16x8*>(w1r0 + k0 + boff);
        b.h[1] = *reinterpret_cast<const bf16x8*>(w1r0 + k0 + boff + 8);
        WMMA_BF16(acc0a, alo.v, b.v);
        WMMA_BF16(acc0b, ahi.v, b.v);
        b.h[0] = *reinterpret_cast<const bf16x8*>(w1r1 + k0 + boff);
        b.h[1] = *reinterpret_cast<const bf16x8*>(w1r1 + k0 + boff + 8);
        WMMA_BF16(acc1a, alo.v, b.v);
        WMMA_BF16(acc1b, ahi.v, b.v);
        b.h[0] = *reinterpret_cast<const bf16x8*>(w1r2 + k0 + boff);
        b.h[1] = *reinterpret_cast<const bf16x8*>(w1r2 + k0 + boff + 8);
        WMMA_BF16(acc2a, alo.v, b.v);
        WMMA_BF16(acc2b, ahi.v, b.v);
        b.h[0] = *reinterpret_cast<const bf16x8*>(w1r3 + k0 + boff);
        b.h[1] = *reinterpret_cast<const bf16x8*>(w1r3 + k0 + boff + 8);
        WMMA_BF16(acc3a, alo.v, b.v);
        WMMA_BF16(acc3b, ahi.v, b.v);

        if constexpr (HAS_A2) {
            FragAB a2l, a2h;
            a2l.h[0] = *reinterpret_cast<const bf16x8*>(a2lo + k0 + aoff0);
            a2l.h[1] = *reinterpret_cast<const bf16x8*>(a2lo + k0 + aoff1);
            a2h.h[0] = *reinterpret_cast<const bf16x8*>(a2hi + k0 + aoff0);
            a2h.h[1] = *reinterpret_cast<const bf16x8*>(a2hi + k0 + aoff1);

            b.h[0] = *reinterpret_cast<const bf16x8*>(w2r0 + k0 + boff);
            b.h[1] = *reinterpret_cast<const bf16x8*>(w2r0 + k0 + boff + 8);
            WMMA_BF16(acc0a, a2l.v, b.v);
            WMMA_BF16(acc0b, a2h.v, b.v);
            b.h[0] = *reinterpret_cast<const bf16x8*>(w2r1 + k0 + boff);
            b.h[1] = *reinterpret_cast<const bf16x8*>(w2r1 + k0 + boff + 8);
            WMMA_BF16(acc1a, a2l.v, b.v);
            WMMA_BF16(acc1b, a2h.v, b.v);
            b.h[0] = *reinterpret_cast<const bf16x8*>(w2r2 + k0 + boff);
            b.h[1] = *reinterpret_cast<const bf16x8*>(w2r2 + k0 + boff + 8);
            WMMA_BF16(acc2a, a2l.v, b.v);
            WMMA_BF16(acc2b, a2h.v, b.v);
            b.h[0] = *reinterpret_cast<const bf16x8*>(w2r3 + k0 + boff);
            b.h[1] = *reinterpret_cast<const bf16x8*>(w2r3 + k0 + boff + 8);
            WMMA_BF16(acc3a, a2l.v, b.v);
            WMMA_BF16(acc3b, a2h.v, b.v);
        }
    }

    // Epilogue: bias + tiny K=5 input projection + LSTM nonlinearity.
    const float bi = bias[0 * HID + j];
    const float bf = bias[1 * HID + j];
    const float bg = bias[2 * HID + j];
    const float bo = bias[3 * HID + j];

    float wxi[OUTD], wxf[OUTD], wxg[OUTD], wxo[OUTD];
    const bool hasx = (xin != nullptr);
    if (hasx) {
        #pragma unroll
        for (int k = 0; k < OUTD; ++k) {
            wxi[k] = Wx[(size_t)(0 * HID + j) * OUTD + k];
            wxf[k] = Wx[(size_t)(1 * HID + j) * OUTD + k];
            wxg[k] = Wx[(size_t)(2 * HID + j) * OUTD + k];
            wxo[k] = Wx[(size_t)(3 * HID + j) * OUTD + k];
        }
    }

    // C/D layout: VGPR r -> M = r (lanes 0-15) or r+8 (lanes 16-31), N = lane&15
    #pragma unroll
    for (int mt = 0; mt < 2; ++mt) {
        const v8f& g0 = mt ? acc0b : acc0a;
        const v8f& g1 = mt ? acc1b : acc1a;
        const v8f& g2 = mt ? acc2b : acc2a;
        const v8f& g3 = mt ? acc3b : acc3a;
        #pragma unroll
        for (int r = 0; r < 8; ++r) {
            const int m = mbase + mt * 16 + (half << 3) + r;
            float gi = g0[r] + bi;
            float gf = g1[r] + bf;
            float gg = g2[r] + bg;
            float go = g3[r] + bo;
            if (hasx) {
                const float* xr = xin + (size_t)m * xstride;
                #pragma unroll
                for (int k = 0; k < OUTD; ++k) {
                    const float xv = xr[k];
                    gi += xv * wxi[k];
                    gf += xv * wxf[k];
                    gg += xv * wxg[k];
                    go += xv * wxo[k];
                }
            }
            const float iv = sigmoidf(gi);
            const float fv = sigmoidf(gf);
            const float gv = tanhf(gg);
            const float ov = sigmoidf(go);
            const size_t off = (size_t)m * HID + j;
            const float c2 = fv * c_in[off] + iv * gv;
            c_out[off] = c2;
            h_out[off] = (__bf16)(ov * tanhf(c2));
        }
    }
}

// ---------------------------------------------------------------------------
// y[m, d] = h1[m, :] . Wout[d, :] + bout[d]   (K=1024, D=5) -> d_out[m, t, d]
// One wave per (m, d): lane-parallel partial dot + __shfl_xor reduction.
// ---------------------------------------------------------------------------
__global__ void out_proj_kernel(const __bf16* __restrict__ h1, const __bf16* __restrict__ Wout,
                                const float* __restrict__ bout, float* __restrict__ ybase)
{
    const int wid  = (blockIdx.x * blockDim.x + threadIdx.x) >> 5;   // wave id
    const int lane = threadIdx.x & 31;
    if (wid >= BATCH * OUTD) return;
    const int m = wid / OUTD;
    const int d = wid % OUTD;
    const __bf16* hr = h1 + (size_t)m * HID + lane * 8;
    const __bf16* wr = Wout + (size_t)d * HID + lane * 8;
    float s = 0.0f;
    #pragma unroll
    for (int k = 0; k < HID; k += 32 * 8) {      // 4 iterations of 8 elems/lane
        const bf16x8 hv = *reinterpret_cast<const bf16x8*>(hr + k);
        const bf16x8 wv = *reinterpret_cast<const bf16x8*>(wr + k);
        #pragma unroll
        for (int e = 0; e < 8; ++e) s += (float)hv[e] * (float)wv[e];
    }
    #pragma unroll
    for (int off = 16; off > 0; off >>= 1) s += __shfl_xor(s, off, 32);
    if (lane == 0) ybase[(size_t)m * (SEQT * OUTD) + d] = s + bout[d];
}

// ---------------------------------------------------------------------------
// h,c init: (z @ W^T + b).reshape(L, B, H)
// ---------------------------------------------------------------------------
__global__ void init_state_kernel(const float* __restrict__ z,
                                  const float* __restrict__ Wh, const float* __restrict__ bh,
                                  const float* __restrict__ Wc, const float* __restrict__ bc,
                                  __bf16* __restrict__ h0, float* __restrict__ c0,
                                  __bf16* __restrict__ h1, float* __restrict__ c1)
{
    const int idx = blockIdx.x * blockDim.x + threadIdx.x;
    if (idx >= BATCH * NLAY * HID) return;
    const int col = idx % (NLAY * HID);
    const int b   = idx / (NLAY * HID);
    const float* zr  = z  + (size_t)b * LATD;
    const float* whr = Wh + (size_t)col * LATD;
    const float* wcr = Wc + (size_t)col * LATD;
    float sh = bh[col];
    float sc = bc[col];
    for (int k = 0; k < LATD; ++k) {
        const float zv = zr[k];
        sh += zv * whr[k];
        sc += zv * wcr[k];
    }
    const int l = col / HID;
    const int jj = col % HID;
    const size_t off = (size_t)b * HID + jj;
    if (l == 0) { h0[off] = (__bf16)sh; c0[off] = sc; }
    else        { h1[off] = (__bf16)sh; c1[off] = sc; }
}

__global__ void f32_to_bf16_kernel(const float* __restrict__ src, __bf16* __restrict__ dst, int n)
{
    const int i = blockIdx.x * blockDim.x + threadIdx.x;
    if (i < n) dst[i] = (__bf16)src[i];
}

__global__ void add_vec_kernel(const float* __restrict__ a, const float* __restrict__ b,
                               float* __restrict__ dst, int n)
{
    const int i = blockIdx.x * blockDim.x + threadIdx.x;
    if (i < n) dst[i] = a[i] + b[i];
}

// ---------------------------------------------------------------------------
extern "C" void kernel_launch(void* const* d_in, const int* in_sizes, int n_in,
                              void* d_out, int out_size, void* d_ws, size_t ws_size,
                              hipStream_t stream)
{
    (void)in_sizes; (void)n_in; (void)out_size; (void)ws_size;

    const float* z    = (const float*)d_in[0];
    const float* Wihz = (const float*)d_in[1];
    const float* bihz = (const float*)d_in[2];
    const float* Wicz = (const float*)d_in[3];
    const float* bicz = (const float*)d_in[4];
    const float* Wih0 = (const float*)d_in[5];
    const float* Whh0 = (const float*)d_in[6];
    const float* bih0 = (const float*)d_in[7];
    const float* bhh0 = (const float*)d_in[8];
    const float* Wih1 = (const float*)d_in[9];
    const float* Whh1 = (const float*)d_in[10];
    const float* bih1 = (const float*)d_in[11];
    const float* bhh1 = (const float*)d_in[12];
    const float* Wout = (const float*)d_in[13];
    const float* bout = (const float*)d_in[14];
    float* out = (float*)d_out;

    // Workspace carve-out (~34 MB)
    char* ws = (char*)d_ws;
    size_t off = 0;
    auto carve = [&](size_t bytes) -> char* {
        char* p = ws + off;
        off += (bytes + 255) & ~(size_t)255;
        return p;
    };
    __bf16* Whh0b = (__bf16*)carve((size_t)4 * HID * HID * sizeof(__bf16));
    __bf16* Wih1b = (__bf16*)carve((size_t)4 * HID * HID * sizeof(__bf16));
    __bf16* Whh1b = (__bf16*)carve((size_t)4 * HID * HID * sizeof(__bf16));
    __bf16* Woutb = (__bf16*)carve((size_t)OUTD * HID * sizeof(__bf16));
    float*  b0    = (float*)carve((size_t)4 * HID * sizeof(float));
    float*  b1    = (float*)carve((size_t)4 * HID * sizeof(float));
    __bf16* h0buf[2];
    __bf16* h1buf[2];
    h0buf[0] = (__bf16*)carve((size_t)BATCH * HID * sizeof(__bf16));
    h0buf[1] = (__bf16*)carve((size_t)BATCH * HID * sizeof(__bf16));
    h1buf[0] = (__bf16*)carve((size_t)BATCH * HID * sizeof(__bf16));
    h1buf[1] = (__bf16*)carve((size_t)BATCH * HID * sizeof(__bf16));
    float* c0 = (float*)carve((size_t)BATCH * HID * sizeof(float));
    float* c1 = (float*)carve((size_t)BATCH * HID * sizeof(float));

    // One-time prep: bf16 weight conversion, fused biases, state init.
    {
        const int nbig = 4 * HID * HID;
        f32_to_bf16_kernel<<<(nbig + 255) / 256, 256, 0, stream>>>(Whh0, Whh0b, nbig);
        f32_to_bf16_kernel<<<(nbig + 255) / 256, 256, 0, stream>>>(Wih1, Wih1b, nbig);
        f32_to_bf16_kernel<<<(nbig + 255) / 256, 256, 0, stream>>>(Whh1, Whh1b, nbig);
        const int nout = OUTD * HID;
        f32_to_bf16_kernel<<<(nout + 255) / 256, 256, 0, stream>>>(Wout, Woutb, nout);
        const int nb = 4 * HID;
        add_vec_kernel<<<(nb + 255) / 256, 256, 0, stream>>>(bih0, bhh0, b0, nb);
        add_vec_kernel<<<(nb + 255) / 256, 256, 0, stream>>>(bih1, bhh1, b1, nb);
        const int ni = BATCH * NLAY * HID;
        init_state_kernel<<<(ni + 255) / 256, 256, 0, stream>>>(z, Wihz, bihz, Wicz, bicz,
                                                               h0buf[0], c0, h1buf[0], c1);
    }

    // Autoregressive decode: 256 sequential steps.
    const dim3 cell_grid(BATCH / 64, HID / 64);
    const dim3 cell_block(256);
    const dim3 yp_grid((BATCH * OUTD * 32 + 255) / 256);   // one wave per (m,d)
    for (int t = 0; t < SEQT; ++t) {
        const __bf16* h0in = h0buf[t & 1];
        __bf16*       h0o  = h0buf[(t + 1) & 1];
        const __bf16* h1in = h1buf[t & 1];
        __bf16*       h1o  = h1buf[(t + 1) & 1];
        // x_t = y_{t-1}, read straight from d_out (row stride T*D); zeros at t=0.
        const float* xin = (t == 0) ? nullptr : (out + (size_t)(t - 1) * OUTD);

        lstm_cell_wmma<false><<<cell_grid, cell_block, 0, stream>>>(
            h0in, Whh0b, nullptr, nullptr, xin, SEQT * OUTD, Wih0, b0, c0, c0, h0o);

        lstm_cell_wmma<true><<<cell_grid, cell_block, 0, stream>>>(
            h0o, Wih1b, h1in, Whh1b, nullptr, 0, nullptr, b1, c1, c1, h1o);

        out_proj_kernel<<<yp_grid, 256, 0, stream>>>(h1o, Woutb, bout, out + (size_t)t * OUTD);
    }
}
